// log_PINN_32667521253624
// MI455X (gfx1250) — matrix operand order repaired and checked by reading, not compiled, not run on hardware
//
#include <hip/hip_runtime.h>
#include <hip/hip_bf16.h>

typedef __attribute__((ext_vector_type(16))) _Float16 v16h;
typedef __attribute__((ext_vector_type(8)))  _Float16 v8h;
typedef __attribute__((ext_vector_type(8)))  float    v8f;

#define HDIM  256
#define NCOMP 15   // jet: 1 value + 4 first derivs + 10 second derivs (i<=j)
#define NC16  16   // padded comp count so 8 waves tile 16 comps x 2 n-subtiles evenly

// ---------------------------------------------------------------------------
// Convert f32 weight [k][n] -> f16 transposed [n][k] so WMMA B-fragments are
// two contiguous 16B loads per lane.
// ---------------------------------------------------------------------------
__global__ __launch_bounds__(256) void k_cvt_transpose(const float* __restrict__ src,
                                                       _Float16* __restrict__ dst) {
  int idx = blockIdx.x * 256 + threadIdx.x;   // 65536 threads
  int k = idx >> 8, n = idx & 255;
  dst[(size_t)n * HDIM + k] = (_Float16)src[(size_t)k * HDIM + n];
}

// ---------------------------------------------------------------------------
// Layer 1 (4 -> 256) jet + tanh. z = x@W1+b1; dz_i = W1[i][n]; d2z = 0.
// Output planes: H[c * B*256 + s*256 + n], f16 (planes 0..14; plane 15 unused).
// ---------------------------------------------------------------------------
__global__ __launch_bounds__(256) void k_layer1_jet(const float* __restrict__ x,
    const float* __restrict__ W1, const float* __restrict__ b1,
    _Float16* __restrict__ H, int B) {
  int idx = blockIdx.x * 256 + threadIdx.x;
  int n = idx & 255, s = idx >> 8;
  if (s >= B) return;
  float w[4], z = b1[n];
#pragma unroll
  for (int i = 0; i < 4; ++i) { w[i] = W1[i * HDIM + n]; z += x[s * 4 + i] * w[i]; }
  float a = tanhf(z), ap = 1.f - a * a, app = -2.f * a * ap;
  size_t plane = (size_t)B * HDIM, base = (size_t)s * HDIM + n;
  H[base] = (_Float16)a;
#pragma unroll
  for (int i = 0; i < 4; ++i) H[(1 + i) * plane + base] = (_Float16)(ap * w[i]);
  int c = 5;
#pragma unroll
  for (int i = 0; i < 4; ++i)
#pragma unroll
    for (int j = i; j < 4; ++j) { H[c * plane + base] = (_Float16)(app * w[i] * w[j]); ++c; }
}

// ---------------------------------------------------------------------------
// Fused hidden layer: jet-GEMM (WMMA f16 -> f32) + bias + tanh-jet, Z kept in
// LDS only. Block = 16 samples x 32 neurons x all 16 (padded) jet components.
// Wave w: n-subtile (w&1), comps {w>>1, +4, +8, +12}; B fragment reused x4.
// ---------------------------------------------------------------------------
__global__ __launch_bounds__(256) void k_layer_fused(const _Float16* __restrict__ A,
    const _Float16* __restrict__ Wt, const float* __restrict__ bias,
    _Float16* __restrict__ H, int B) {
  __shared__ float Zs[NC16][16][32];          // 32 KB
  int lane = threadIdx.x & 31;
  int wid  = threadIdx.x >> 5;
  int ng = blockIdx.x & 7;                    // neuron group of 32
  int sg = blockIdx.x >> 3;                   // sample group of 16
  int nt = wid & 1;                           // 16-col subtile
  int cb = wid >> 1;                          // comp base (0..3)
  size_t plane = (size_t)B * HDIM;
  int l15 = lane & 15;
  int alo = (lane >> 4) * 16;                 // A frag K-offset per lane half
  int blo = (lane >> 4) * 16;                 // B frag K-offset per lane half

  const _Float16* bpt = Wt + (size_t)(ng * 32 + nt * 16 + l15) * HDIM + blo;
  const _Float16* apt = A + ((size_t)cb * B + (size_t)sg * 16 + l15) * HDIM + alo;
  const size_t astep = (size_t)4 * B * HDIM;  // +4 component planes (in rows)

  v8f acc[4] = {{}, {}, {}, {}};
#pragma unroll
  for (int kb = 0; kb < HDIM; kb += 32) {
    v8h b0 = *(const v8h*)(bpt + kb);
    v8h b1 = *(const v8h*)(bpt + kb + 8);
    v16h bv;
#pragma unroll
    for (int i = 0; i < 8; ++i) { bv[i] = b0[i]; bv[i + 8] = b1[i]; }
#pragma unroll
    for (int q = 0; q < 4; ++q) {
      const _Float16* ap = apt + q * astep + kb;
      v8h a0 = *(const v8h*)(ap);
      v8h a1 = *(const v8h*)(ap + 8);
      v16h av;
#pragma unroll
      for (int i = 0; i < 8; ++i) { av[i] = a0[i]; av[i + 8] = a1[i]; }
      acc[q] = __builtin_amdgcn_wmma_f32_16x16x32_f16(false, av, false, bv,
                                                      (short)0, acc[q], false, false);
    }
  }
  // C/D layout: VGPR j -> M = j + 8*(lane>=16), N = lane&15
  int srow = (lane >> 4) * 8;
  int scol = nt * 16 + l15;
#pragma unroll
  for (int q = 0; q < 4; ++q)
#pragma unroll
    for (int j = 0; j < 8; ++j)
      Zs[cb + 4 * q][srow + j][scol] = acc[q][j];
  __syncthreads();

  // tanh-jet on the 16x32 patch; write f16 activation-jet planes
  for (int e = threadIdx.x; e < 16 * 32; e += 256) {
    int nl = e & 31, sl = e >> 5;
    int n = ng * 32 + nl;
    size_t base = ((size_t)sg * 16 + sl) * HDIM + n;
    float z[NCOMP];
#pragma unroll
    for (int c = 0; c < NCOMP; ++c) z[c] = Zs[c][sl][nl];
    float a = tanhf(z[0] + bias[n]);
    float ap = 1.f - a * a, app = -2.f * a * ap;
    H[base] = (_Float16)a;
#pragma unroll
    for (int i = 0; i < 4; ++i) H[(1 + i) * plane + base] = (_Float16)(ap * z[1 + i]);
    int c = 5;
#pragma unroll
    for (int i = 0; i < 4; ++i)
#pragma unroll
      for (int j = i; j < 4; ++j) {
        H[c * plane + base] = (_Float16)(ap * z[c] + app * z[1 + i] * z[1 + j]);
        ++c;
      }
  }
}

// ---------------------------------------------------------------------------
// Head: per-sample wave (wave32). Reduce f = h3 . Wo + bo for all 15 jet
// components, g_rr = exp(f) jet, then the diagonal-metric Einstein tensor.
// ---------------------------------------------------------------------------
__global__ __launch_bounds__(256) void k_head_einstein(const _Float16* __restrict__ H3,
    const float* __restrict__ Wo, const float* __restrict__ bo,
    const float* __restrict__ coords, float* __restrict__ out, int B) {
  int lane = threadIdx.x & 31;
  int s = blockIdx.x * 8 + (threadIdx.x >> 5);
  if (s >= B) return;
  size_t plane = (size_t)B * HDIM, base = (size_t)s * HDIM;
  float acc[NCOMP];
#pragma unroll
  for (int c = 0; c < NCOMP; ++c) acc[c] = 0.f;
  for (int n = lane; n < HDIM; n += 32) {
    float wv = Wo[n];
#pragma unroll
    for (int c = 0; c < NCOMP; ++c) acc[c] += wv * (float)H3[c * plane + base + n];
  }
#pragma unroll
  for (int off = 16; off > 0; off >>= 1)
#pragma unroll
    for (int c = 0; c < NCOMP; ++c) acc[c] += __shfl_xor(acc[c], off, 32);
  if (lane != 0) return;

  // ---- jet of scalar f, then g_rr = exp(f) ----
  float f0 = acc[0] + bo[0];
  float fd[4] = {acc[1], acc[2], acc[3], acc[4]};
  float f2[4][4];
  { int c = 5;
    for (int i = 0; i < 4; ++i)
      for (int j = i; j < 4; ++j) { f2[i][j] = f2[j][i] = acc[c]; ++c; } }
  float G = expf(f0);
  float Gd[4];
  for (int i = 0; i < 4; ++i) Gd[i] = G * fd[i];
  float Gdd[4][4];
  for (int i = 0; i < 4; ++i)
    for (int j = 0; j < 4; ++j) Gdd[i][j] = G * (f2[i][j] + fd[i] * fd[j]);

  // ---- diagonal metric g = diag(-1, G, r^2, r^2 sin^2 th) and derivatives ----
  float r = coords[(size_t)s * 4 + 1], th = coords[(size_t)s * 4 + 2];
  float sn = sinf(th), cs = cosf(th);
  float ginv[4] = {-1.f, 1.f / G, 1.f / (r * r), 1.f / (r * r * sn * sn)};
  float dg[4][4] = {};          // dg[j][k] = d_k g_jj
  for (int k = 0; k < 4; ++k) dg[1][k] = Gd[k];
  dg[2][1] = 2.f * r;
  dg[3][1] = 2.f * r * sn * sn;
  dg[3][2] = 2.f * r * r * sn * cs;
  float d2g[4][4][4] = {};      // d2g[j][k][l] = d_k d_l g_jj
  for (int k = 0; k < 4; ++k)
    for (int l = 0; l < 4; ++l) d2g[1][k][l] = Gdd[k][l];
  d2g[2][1][1] = 2.f;
  d2g[3][1][1] = 2.f * sn * sn;
  d2g[3][1][2] = d2g[3][2][1] = 4.f * r * sn * cs;
  d2g[3][2][2] = 2.f * r * r * (cs * cs - sn * sn);

  // ---- Christoffel: Gam^a_bc = 0.5 g^aa (d_c g_ab + d_b g_ac - d_a g_bc) ----
  float gam[4][4][4];
  for (int a = 0; a < 4; ++a)
    for (int bb = 0; bb < 4; ++bb)
      for (int cc = 0; cc < 4; ++cc) {
        float sym = 0.f;
        if (a == bb)  sym += dg[a][cc];
        if (a == cc)  sym += dg[a][bb];
        if (bb == cc) sym -= dg[bb][a];
        gam[a][bb][cc] = 0.5f * ginv[a] * sym;
      }
  auto dGam = [&](int a, int bb, int cc, int k) -> float {
    float sym = 0.f, dsym = 0.f;
    if (a == bb)  { sym += dg[a][cc]; dsym += d2g[a][cc][k]; }
    if (a == cc)  { sym += dg[a][bb]; dsym += d2g[a][bb][k]; }
    if (bb == cc) { sym -= dg[bb][a]; dsym -= d2g[bb][a][k]; }
    return 0.5f * (-dg[a][k] * ginv[a] * ginv[a] * sym + ginv[a] * dsym);
  };

  float ricci[4][4];
  for (int bb = 0; bb < 4; ++bb)
    for (int dd = 0; dd < 4; ++dd) {
      float rc = 0.f;
      for (int a = 0; a < 4; ++a)
        rc += dGam(a, dd, bb, a) - dGam(a, a, bb, dd);
      for (int a = 0; a < 4; ++a)
        for (int e = 0; e < 4; ++e)
          rc += gam[a][a][e] * gam[e][dd][bb] - gam[a][dd][e] * gam[e][a][bb];
      ricci[bb][dd] = rc;
    }
  float RS = 0.f;
  for (int bb = 0; bb < 4; ++bb) RS += ginv[bb] * ricci[bb][bb];
  for (int a = 0; a < 4; ++a)
    for (int bb = 0; bb < 4; ++bb) {
      float e = ginv[a] * ginv[bb] * ricci[a][bb];
      if (a == bb) e -= 0.5f * ginv[a] * RS;
      out[(size_t)s * 16 + a * 4 + bb] = e;
    }
}

// ---------------------------------------------------------------------------
extern "C" void kernel_launch(void* const* d_in, const int* in_sizes, int n_in,
                              void* d_out, int out_size, void* d_ws, size_t ws_size,
                              hipStream_t stream) {
  const float* coords = (const float*)d_in[0];
  const float* W1 = (const float*)d_in[1];
  const float* b1 = (const float*)d_in[2];
  const float* W2 = (const float*)d_in[3];
  const float* b2 = (const float*)d_in[4];
  const float* W3 = (const float*)d_in[5];
  const float* b3 = (const float*)d_in[6];
  const float* Wo = (const float*)d_in[7];
  const float* bo = (const float*)d_in[8];
  const int B = in_sizes[0] / 4;                 // 8192
  const size_t plane = (size_t)B * HDIM;

  char* ws = (char*)d_ws;
  _Float16* W2h = (_Float16*)ws;                              // 64K halves
  _Float16* W3h = W2h + (size_t)HDIM * HDIM;                  // 64K halves
  _Float16* HA = (_Float16*)(ws + 2 * (size_t)HDIM * HDIM * sizeof(_Float16));
  _Float16* HB = HA + (size_t)NC16 * plane;      // 16 padded planes each

  const int elem_blocks  = (B * HDIM) / 256;     // 8192
  const int fused_blocks = (B / 16) * 8;         // 4096: 16-sample x 32-neuron patches

  k_cvt_transpose<<<(HDIM * HDIM) / 256, 256, 0, stream>>>(W2, W2h);
  k_cvt_transpose<<<(HDIM * HDIM) / 256, 256, 0, stream>>>(W3, W3h);

  k_layer1_jet<<<elem_blocks, 256, 0, stream>>>(coords, W1, b1, HA, B);
  k_layer_fused<<<fused_blocks, 256, 0, stream>>>(HA, W2h, b2, HB, B);
  k_layer_fused<<<fused_blocks, 256, 0, stream>>>(HB, W3h, b3, HA, B);
  k_head_einstein<<<B / 8, 256, 0, stream>>>(HA, Wo, bo, coords, (float*)d_out, B);
}
